// Net_42176578846907
// MI455X (gfx1250) — matrix operand order repaired and checked by reading, not compiled, not run on hardware
//
#include <hip/hip_runtime.h>
#include <math.h>

// CDNA5 / gfx1250: wave32, WMMA 16x16x32 f16 -> f32 accumulate.
typedef __attribute__((ext_vector_type(16))) _Float16 v16h;
typedef __attribute__((ext_vector_type(8)))  float    v8f;

#define DIM 128   // input feature dim (K)
#define NQ  10    // number of qubits (N, padded to 16 in the WMMA tile)

__device__ __forceinline__ void pack4(v16h& v, int i, float4 f) {
  v[i + 0] = (_Float16)f.x;
  v[i + 1] = (_Float16)f.y;
  v[i + 2] = (_Float16)f.z;
  v[i + 3] = (_Float16)f.w;
}

// Kernel 1: theta[b, q] = atan( sum_k x[b,k] * W[q,k] + bias[q] )
// One wave32 per 16-row tile; 8 waves (8 tiles) per 256-thread block.
// K=128 -> 4 chained v_wmma_f32_16x16x32_f16 ops per tile.
__global__ __launch_bounds__(256) void theta_wmma_kernel(
    const float* __restrict__ x, const float* __restrict__ W,
    const float* __restrict__ bias, float* __restrict__ theta, int ntiles)
{
  const int lane = threadIdx.x & 31;
  const int wave = threadIdx.x >> 5;
  const int tile = blockIdx.x * 8 + wave;
  if (tile >= ntiles) return;           // uniform per wave: EXEC stays all-ones

  const int hi = lane >> 4;             // 0: lanes 0-15, 1: lanes 16-31
  const int q  = lane & 15;             // B/C/D column held by this lane
  const int wq = (q < NQ) ? q : 0;      // clamp dead columns into valid memory

  // ---- B fragments: B[k][n] = W[n][k], 4 K-steps of 32 ----
  // ISA layout (16-bit B 32x16): lane n holds column n%16;
  // halves 0-7 -> K=base..base+7, halves 8-15 -> K=base+16..base+23, base = (lane>=16 ? 8 : 0).
  v16h bfrag[4];
  {
    const float* wr = W + (size_t)wq * DIM + hi * 8;
#pragma unroll
    for (int s = 0; s < 4; ++s) {
      const float4* wp = (const float4*)(wr + s * 32);
      pack4(bfrag[s], 0,  wp[0]);
      pack4(bfrag[s], 4,  wp[1]);
      pack4(bfrag[s], 8,  wp[4]);
      pack4(bfrag[s], 12, wp[5]);
    }
  }

  // ---- A fragments: row = tile*16 + lane%16 ----
  // ISA layout (16-bit A 16x32): halves 0-7 -> K=base..base+7,
  // halves 8-15 -> K=base+16..base+23, base = (lane>=16 ? 8 : 0).
  const int row = tile * 16 + (lane & 15);
  const float* xr = x + (size_t)row * DIM + hi * 8;

  v8f acc = {};
#pragma unroll
  for (int s = 0; s < 4; ++s) {
    const float* ap = xr + s * 32;
    v16h a;
    pack4(a, 0,  *(const float4*)(ap + 0));
    pack4(a, 4,  *(const float4*)(ap + 4));
    pack4(a, 8,  *(const float4*)(ap + 16));
    pack4(a, 12, *(const float4*)(ap + 20));
    // (neg_a, A, neg_b, B, c_mod, C, reuse_a, reuse_b)
    acc = __builtin_amdgcn_wmma_f32_16x16x32_f16(
        false, a, false, bfrag[s], (short)0, acc, false, false);
  }

  // ---- C/D layout: lane holds column q; VGPR r holds row rbase + r ----
  if (q < NQ) {
    const float bq    = bias[q];
    const int   rbase = tile * 16 + hi * 8;
#pragma unroll
    for (int r = 0; r < 8; ++r) {
      theta[(size_t)(rbase + r) * NQ + q] = atanf(acc[r] + bq);
    }
  }
}

// Kernel 2: analytically-reduced circuit.
// After H, the state is |+>^10; RX(theta) only adds a phase on |+>; the CNOT
// ring is the identity on |+...+>. All 1024 probabilities equal
// P = (1/1024) * prod_q (cos^2 + sin^2), and <Z_q> = 512P - 512P == 0.0
// bit-exactly (kept as a runtime subtraction so NaN inputs propagate like the
// reference would).
__global__ __launch_bounds__(256) void expz_kernel(
    const float* __restrict__ theta, float* __restrict__ out, int batch)
{
  const int i = blockIdx.x * blockDim.x + threadIdx.x;
  if (i >= batch) return;
  const float* t = theta + (size_t)i * NQ;

  float p = 1.0f / 1024.0f;
#pragma unroll
  for (int q = 0; q < NQ; ++q) {
    const float h = 0.5f * t[q];
    const float c = __cosf(h);
    const float s = __sinf(h);
    p *= (c * c + s * s);
  }
  const float p0 = 512.0f * p;   // prob(bit_q = 0)
  const float p1 = 512.0f * p;   // prob(bit_q = 1)
  const float z  = p0 - p1;      // exactly 0.0f for finite theta

#pragma unroll
  for (int q = 0; q < NQ; ++q) out[(size_t)i * NQ + q] = z;
}

extern "C" void kernel_launch(void* const* d_in, const int* in_sizes, int n_in,
                              void* d_out, int out_size, void* d_ws, size_t ws_size,
                              hipStream_t stream)
{
  const float* x    = (const float*)d_in[0];   // (batch, 128)
  const float* W    = (const float*)d_in[1];   // (10, 128)
  const float* bias = (const float*)d_in[2];   // (10,)
  float* out   = (float*)d_out;                // (batch, 10)
  float* theta = (float*)d_ws;                 // batch*10 floats of scratch

  const int batch  = in_sizes[0] / DIM;        // 32768
  const int ntiles = (batch + 15) / 16;        // 2048
  const int blocks1 = (ntiles + 7) / 8;        // 8 tiles (8 waves) per block
  theta_wmma_kernel<<<blocks1, 256, 0, stream>>>(x, W, bias, theta, ntiles);

  const int blocks2 = (batch + 255) / 256;
  expz_kernel<<<blocks2, 256, 0, stream>>>(theta, out, batch);
}
